// AngularPenaltySMLoss_91010357002394
// MI455X (gfx1250) — matrix operand order repaired and checked by reading, not compiled, not run on hardware
//
#include <hip/hip_runtime.h>
#include <math.h>

// ---------------- problem constants (from reference) ----------------
#define NROWS   4096
#define DIM     512
#define NCLS    50000
#define S_SCALE 30.0f
#define MARGIN  0.4f
#define EPS_C   1e-7f

#define CTILE   64              // classes per workgroup (LDS-staged W tile)
#define NCTILE  ((NCLS + CTILE - 1) / CTILE)   // 782
#define MTILE   128             // rows per workgroup iteration (4 waves x 32)

typedef __attribute__((ext_vector_type(16))) __bf16 v16bf;
typedef __attribute__((ext_vector_type(8)))  __bf16 v8bf;
typedef __attribute__((ext_vector_type(8)))  float  v8f;

union ABFrag { v16bf v; v8bf h[2]; };

// =====================================================================
// Kernel 1: L2-normalize rows of x -> bf16; zero the per-row accumulators
// one wave32 per row, 8 waves per block
// =====================================================================
__global__ __launch_bounds__(256) void k_normalize(
    const float* __restrict__ x, __bf16* __restrict__ xn,
    float* __restrict__ excl, float* __restrict__ tgtcos)
{
    const int wave = threadIdx.x >> 5;
    const int lane = threadIdx.x & 31;
    const int row  = blockIdx.x * 8 + wave;
    const float* xr = x + (size_t)row * DIM;

    float vals[16];
    float ss = 0.f;
#pragma unroll
    for (int j = 0; j < 4; ++j) {
        float4 v = *(const float4*)(xr + lane * 16 + j * 4);
        vals[j*4+0] = v.x; vals[j*4+1] = v.y; vals[j*4+2] = v.z; vals[j*4+3] = v.w;
        ss += v.x*v.x + v.y*v.y + v.z*v.z + v.w*v.w;
    }
#pragma unroll
    for (int m = 1; m < 32; m <<= 1) ss += __shfl_xor(ss, m);
    const float rn = 1.0f / sqrtf(ss);

    __bf16* dst = xn + (size_t)row * DIM + lane * 16;
#pragma unroll
    for (int i = 0; i < 16; ++i) dst[i] = (__bf16)(vals[i] * rn);

    if (lane == 0) { excl[row] = 0.f; tgtcos[row] = 0.f; }
}

// =====================================================================
// Kernel 2: fused GEMM (xn @ W^T) + masked exp-sum epilogue.
// 1 workgroup (128 thr = 4 waves) per 64-class tile; W tile staged to
// LDS as bf16 once, reused across all M tiles (W read from HBM once).
// Each wave: 32(M) x 64(N) strip -> 8x v_wmma_f32_16x16x32_bf16 / K-step,
// each B fragment (ds_load) feeds TWO WMMAs (halved LDS traffic per MAC).
// =====================================================================
__global__ __launch_bounds__(128) void k_gemm_arcface(
    const float* __restrict__ W, const __bf16* __restrict__ xn,
    const int* __restrict__ target,
    float* __restrict__ excl, float* __restrict__ tgtcos)
{
    __shared__ __bf16 ldsW[CTILE * DIM];      // 64 KB of the 320 KB WGP LDS

    const int tid = threadIdx.x;
    const int c0  = blockIdx.x * CTILE;

    // ---- stage W[c0:c0+64, :] -> LDS, fp32 -> bf16, zero-pad past NCLS ----
#pragma unroll 4
    for (int i = 0; i < 64; ++i) {            // 64 float4 per thread
        const int e4   = tid + i * 128;       // float4 index in tile (0..8191)
        const int rowc = e4 >> 7;             // 128 float4 per class row
        const int k4   = (e4 & 127) * 4;
        const int c    = c0 + rowc;
        float4 w4 = make_float4(0.f, 0.f, 0.f, 0.f);
        if (c < NCLS) w4 = *(const float4*)(W + (size_t)c * DIM + k4);
        __bf16* d = &ldsW[rowc * DIM + k4];
        d[0] = (__bf16)w4.x; d[1] = (__bf16)w4.y;
        d[2] = (__bf16)w4.z; d[3] = (__bf16)w4.w;
    }
    __syncthreads();

    const int wave = tid >> 5;
    const int lane = tid & 31;
    const int half = lane >> 4;               // which 16-lane half
    const int lr   = lane & 15;

    for (int mt = 0; mt < NROWS / MTILE; ++mt) {
        const int m0 = mt * MTILE + wave * 32;    // this wave's 32 rows
        v8f acc[2][4] = {};                       // 2 M-subtiles x 4 N-subtiles

        const __bf16* arow0 = xn + (size_t)(m0 + lr) * DIM;        // rows m0..m0+15
        const __bf16* arow1 = xn + (size_t)(m0 + 16 + lr) * DIM;   // rows m0+16..m0+31

#pragma unroll 2
        for (int k0 = 0; k0 < DIM; k0 += 32) {
            // A fragments (16x32 bf16): half 0 -> K {k0..k0+15},
            //                           half 1 -> K {+16..+31}
            ABFrag a0, a1;
            a0.h[0] = *(const v8bf*)(arow0 + k0 + half * 16);
            a0.h[1] = *(const v8bf*)(arow0 + k0 + half * 16 + 8);
            a1.h[0] = *(const v8bf*)(arow1 + k0 + half * 16);
            a1.h[1] = *(const v8bf*)(arow1 + k0 + half * 16 + 8);

            // B fragments (32x16 bf16): lane = column, contiguous 16 K values
            const __bf16* bbase = &ldsW[k0 + half * 16];
            v16bf b[4];
#pragma unroll
            for (int t = 0; t < 4; ++t)
                b[t] = *(const v16bf*)(bbase + (size_t)(lr + 16 * t) * DIM);

#pragma unroll
            for (int t = 0; t < 4; ++t) {
                acc[0][t] = __builtin_amdgcn_wmma_f32_16x16x32_bf16(
                    false, a0.v, false, b[t], (short)0, acc[0][t], false, false);
                acc[1][t] = __builtin_amdgcn_wmma_f32_16x16x32_bf16(
                    false, a1.v, false, b[t], (short)0, acc[1][t], false, false);
            }
        }

        // ---- epilogue: masked exp-sum per row + target-cosine capture ----
        // C/D layout: VGPR j, lanes 0-15 -> M=j, lanes 16-31 -> M=j+8; N=lr.
#pragma unroll
        for (int ai = 0; ai < 2; ++ai) {
#pragma unroll
            for (int j = 0; j < 8; ++j) {
                const int mrow = m0 + ai * 16 + j + 8 * half;
                const int tg   = target[mrow];
                float rp = 0.f;
#pragma unroll
                for (int t = 0; t < 4; ++t) {
                    const float logit = acc[ai][t][j];
                    const int c = c0 + t * 16 + lr;
                    if (c < NCLS) {
                        if (c == tg) tgtcos[mrow] = logit;   // exactly one writer
                        else         rp += __expf(S_SCALE * logit);
                    }
                }
                // reduce across the 16 lanes of this half (all share mrow)
                rp += __shfl_xor(rp, 1);
                rp += __shfl_xor(rp, 2);
                rp += __shfl_xor(rp, 4);
                rp += __shfl_xor(rp, 8);
                if (lr == 0) atomicAdd(&excl[mrow], rp);     // 1 atomic / row / C-tile
            }
        }
    }
}

// =====================================================================
// Kernel 3: ArcFace scalar epilogue + mean reduction (1 block)
// =====================================================================
__global__ __launch_bounds__(256) void k_final(
    const float* __restrict__ excl, const float* __restrict__ tgtcos,
    float* __restrict__ out)
{
    __shared__ float red[256];
    float acc = 0.f;
    for (int n = threadIdx.x; n < NROWS; n += 256) {
        float t = tgtcos[n];
        t = fminf(fmaxf(t, -1.0f + EPS_C), 1.0f - EPS_C);
        const float num = S_SCALE * __cosf(acosf(t) + MARGIN);
        acc += num - __logf(__expf(num) + excl[n]);
    }
    red[threadIdx.x] = acc;
    __syncthreads();
    for (int s = 128; s > 0; s >>= 1) {
        if (threadIdx.x < s) red[threadIdx.x] += red[threadIdx.x + s];
        __syncthreads();
    }
    if (threadIdx.x == 0) out[0] = -red[0] / (float)NROWS;
}

// =====================================================================
extern "C" void kernel_launch(void* const* d_in, const int* in_sizes, int n_in,
                              void* d_out, int out_size, void* d_ws, size_t ws_size,
                              hipStream_t stream)
{
    const float* x      = (const float*)d_in[0];   // [4096, 512]
    const float* W      = (const float*)d_in[1];   // [50000, 512]
    const int*   target = (const int*)d_in[2];     // [4096]
    float*       out    = (float*)d_out;

    char* ws = (char*)d_ws;
    __bf16* xn     = (__bf16*)ws;                             // 4096*512*2 B
    float*  excl   = (float*)(ws + (size_t)NROWS * DIM * 2);  // 4096 f32
    float*  tgtcos = excl + NROWS;                            // 4096 f32

    k_normalize<<<NROWS / 8, 256, 0, stream>>>(x, xn, excl, tgtcos);
    k_gemm_arcface<<<NCTILE, 128, 0, stream>>>(W, xn, target, excl, tgtcos);
    k_final<<<1, 256, 0, stream>>>(excl, tgtcos, out);
}
